// GatedLinearAttention_75977971466527
// MI455X (gfx1250) — matrix operand order repaired and checked by reading, not compiled, not run on hardware
//
#include <hip/hip_runtime.h>
#include <hip/hip_bf16.h>
#include <cstdint>

// ---------------------------------------------------------------------------
// Gated Linear Attention forward for MI455X (gfx1250), bf16 WMMA everywhere.
//   B=2, T=2048, D=1024, H=4, HDK=256, HDV=512, chunk=64.
// Compute-bound (~80 GFLOP vs ~0.2 GB traffic) -> v_wmma_f32_16x16x32_bf16,
// async global->LDS staging (ASYNCcnt) + double-buffered GEMM tiles.
// ---------------------------------------------------------------------------

#define D_MODEL   1024
#define HEADS     4
#define DK        1024
#define DV        2048
#define HDK       256
#define HDV       512
#define LOW_RANK  16
#define CHUNK     64
#define NCHUNK    32
#define SEQ_T     2048
#define BATCH     2
#define NTOK      (BATCH * SEQ_T)     // 4096 tokens
#define GATE_NORM 16.0f
#define EPS_RMS   1e-5f

typedef __attribute__((ext_vector_type(16))) __bf16 v16bf;
typedef __attribute__((ext_vector_type(8)))  float  v8f;

// GCC-style generic vector to match the builtin's V4i pointee encoding.
typedef int v4i32 __attribute__((vector_size(16)));
typedef __attribute__((address_space(1))) v4i32 as1_v4i;
typedef __attribute__((address_space(3))) v4i32 as3_v4i;

// 16-byte global -> LDS copy on the CDNA5 async path when available.
__device__ __forceinline__ void async_copy16(const void* g, void* l) {
#if __has_builtin(__builtin_amdgcn_global_load_async_to_lds_b128)
  __builtin_amdgcn_global_load_async_to_lds_b128(
      (as1_v4i*)(uintptr_t)g,
      (as3_v4i*)(unsigned)(uintptr_t)l, 0, 0);
#else
  *reinterpret_cast<uint4*>(l) = *reinterpret_cast<const uint4*>(g);
#endif
}

__device__ __forceinline__ void wait_async() {
#if __has_builtin(__builtin_amdgcn_s_wait_asynccnt)
  __builtin_amdgcn_s_wait_asynccnt(0);
#endif
}

__device__ __forceinline__ v8f zero8() {
  v8f z;
#pragma unroll
  for (int i = 0; i < 8; ++i) z[i] = 0.0f;
  return z;
}

// A fragment (16x32 bf16, M x K): lanes 0-15 row M=lane, K 0..15;
// lanes 16-31 row M=lane-16, K 16..31.  p = &A[row][k0] in LDS.
__device__ __forceinline__ v16bf ldfragA(const __bf16* p, int khalf) {
  v16bf a;
  uint4* u = reinterpret_cast<uint4*>(&a);
  u[0] = *reinterpret_cast<const uint4*>(p + 2 * khalf);
  u[1] = *reinterpret_cast<const uint4*>(p + 2 * khalf + 8);
  return a;
}

// B fragment (32x16 bf16, K x N) from K-transposed LDS tile Bt[n][k]:
// lanes 0-15 (col N=lane) hold K=0..15, lanes 16-31 hold K=16..31.
__device__ __forceinline__ v16bf ldfragB(const __bf16* p, int koff) {
  v16bf b;
  uint4* u = reinterpret_cast<uint4*>(&b);
  u[0] = *reinterpret_cast<const uint4*>(p + koff);
  u[1] = *reinterpret_cast<const uint4*>(p + koff + 8);
  return b;
}

__device__ __forceinline__ v8f wmma_bf16(v16bf a, v16bf b, v8f c) {
  return __builtin_amdgcn_wmma_f32_16x16x32_bf16(false, a, false, b,
                                                 (short)0, c, false, false);
}

// ---------------------------------------------------------------------------
// Generic bf16 GEMM: C[M,N] f32 = A[M,K] @ B[K,N], all row-major.
// 128x128 tile per 256-thread block (8 waves in 2x4), BK=64, double-buffered:
// A tile via global_load_async_to_lds_b128, B tile global->regs->LDS after
// compute so global latency hides behind the WMMAs.
// ---------------------------------------------------------------------------
#define BKK 64

__global__ __launch_bounds__(256) void gemm_bf16_kernel(
    const __bf16* __restrict__ A, const __bf16* __restrict__ Bm,
    float* __restrict__ C, int M, int N, int K) {
  __shared__ __bf16 As[2][128 * BKK];   // row-major A tiles
  __shared__ __bf16 Bt[2][128 * BKK];   // K-transposed B tiles: Bt[n][k]
  const int tid  = threadIdx.x;
  const int lane = tid & 31, wave = tid >> 5;
  const int hi = lane >> 4, ll = lane & 15;
  const int khalf = hi ? 8 : 0, koff = hi ? 16 : 0;
  const int bm = blockIdx.y * 128, bn = blockIdx.x * 128;
  const int wm = (wave >> 2) * 64;      // 0 / 64
  const int wn = (wave & 3) * 32;       // 0,32,64,96

  v8f acc[4][2];
#pragma unroll
  for (int mt = 0; mt < 4; ++mt)
#pragma unroll
    for (int nt = 0; nt < 2; ++nt) acc[mt][nt] = zero8();

  const unsigned short* Bg = reinterpret_cast<const unsigned short*>(Bm);
  ushort2 breg[16];   // B tile staged in regs: pairs along k (rows kk, kk+1)

  auto loadB = [&](int k0) {
#pragma unroll
    for (int i = 0; i < 16; ++i) {
      int idx = tid + i * 256;                 // 0..4095
      int kk2 = idx >> 7, n = idx & 127;       // kk = 2*kk2
      size_t g = (size_t)(k0 + 2 * kk2) * N + bn + n;
      breg[i].x = Bg[g];
      breg[i].y = Bg[g + N];
    }
  };
  auto storeB = [&](int buf) {
#pragma unroll
    for (int i = 0; i < 16; ++i) {
      int idx = tid + i * 256;
      int kk2 = idx >> 7, n = idx & 127;
      *reinterpret_cast<ushort2*>(&Bt[buf][n * BKK + 2 * kk2]) = breg[i];
    }
  };
  auto loadA_async = [&](int buf, int k0) {
#pragma unroll
    for (int i = 0; i < 4; ++i) {
      int c = tid + i * 256;                   // 1024 chunks of 16B
      int row = c >> 3, cc = c & 7;
      async_copy16(A + (size_t)(bm + row) * K + k0 + cc * 8,
                   &As[buf][row * BKK + cc * 8]);
    }
  };
  auto compute = [&](int buf) {
#pragma unroll
    for (int ks = 0; ks < 2; ++ks) {
      v16bf b0 = ldfragB(&Bt[buf][(wn + ll) * BKK + ks * 32], koff);
      v16bf b1 = ldfragB(&Bt[buf][(wn + 16 + ll) * BKK + ks * 32], koff);
#pragma unroll
      for (int mt = 0; mt < 4; ++mt) {
        v16bf a = ldfragA(&As[buf][(wm + mt * 16 + ll) * BKK + ks * 32], khalf);
        acc[mt][0] = wmma_bf16(a, b0, acc[mt][0]);
        acc[mt][1] = wmma_bf16(a, b1, acc[mt][1]);
      }
    }
  };

  const int KT = K / BKK;
  loadB(0);
  loadA_async(0, 0);
  storeB(0);
  wait_async();
  __syncthreads();

  for (int kt = 0; kt < KT; ++kt) {
    const int cur = kt & 1, nxt = cur ^ 1;
    if (kt + 1 < KT) {
      loadB((kt + 1) * BKK);                 // global loads in flight
      loadA_async(nxt, (kt + 1) * BKK);      // ASYNCcnt copies in flight
    }
    if (kt + 2 < KT)                          // warm L2 for the tile after next
      __builtin_prefetch(A + (size_t)(bm + (tid >> 1)) * K + (kt + 2) * BKK + (tid & 1) * 32, 0, 1);
    compute(cur);                             // WMMAs overlap the loads above
    if (kt + 1 < KT) storeB(nxt);             // regs -> LDS after compute
    wait_async();
    __syncthreads();
  }

  // C/D layout: m_local = r + 8*(lane>=16), n_local = lane&15
#pragma unroll
  for (int mt = 0; mt < 4; ++mt)
#pragma unroll
    for (int nt = 0; nt < 2; ++nt)
#pragma unroll
      for (int r = 0; r < 8; ++r) {
        int row = bm + wm + mt * 16 + r + hi * 8;
        int col = bn + wn + nt * 16 + ll;
        C[(size_t)row * N + col] = acc[mt][nt][r];
      }
}

// ---------------------------------------------------------------------------
// Elementwise helpers
// ---------------------------------------------------------------------------
__global__ __launch_bounds__(256) void cast_kernel(const float* __restrict__ src,
                                                   __bf16* __restrict__ dst, int n) {
  int i = blockIdx.x * 256 + threadIdx.x;
  if (i < n) dst[i] = (__bf16)src[i];
}

// t1[m, j] = x[m,:] @ Wgk1[:, j]   (4096 x 16, K=1024)
__global__ __launch_bounds__(256) void lr1_kernel(const float* __restrict__ x,
                                                  const float* __restrict__ Wgk1,
                                                  float* __restrict__ t1) {
  int idx = blockIdx.x * 256 + threadIdx.x;     // < NTOK*16
  int m = idx >> 4, j = idx & 15;
  float s = 0.0f;
  for (int kk = 0; kk < D_MODEL; ++kk)
    s += x[(size_t)m * D_MODEL + kk] * Wgk1[kk * LOW_RANK + j];
  t1[idx] = s;
}

// g[m, d] = log_sigmoid(t1[m,:] @ Wgk2[:, d] + bgk2[d]) / GATE_NORM
__global__ __launch_bounds__(256) void gate_kernel(const float* __restrict__ t1,
                                                   const float* __restrict__ Wgk2,
                                                   const float* __restrict__ bgk2,
                                                   float* __restrict__ g) {
  int idx = blockIdx.x * 256 + threadIdx.x;     // < NTOK*DK
  int m = idx >> 10, d = idx & 1023;
  float s = bgk2[d];
#pragma unroll
  for (int r = 0; r < LOW_RANK; ++r)
    s += t1[m * LOW_RANK + r] * Wgk2[r * DK + d];
  float ls = fminf(s, 0.0f) - log1pf(__expf(-fabsf(s)));   // stable log-sigmoid
  g[idx] = ls / GATE_NORM;
}

// In-place cumulative sum of g over t within each 64-long chunk (per column).
__global__ __launch_bounds__(256) void cumsum_kernel(float* __restrict__ g) {
  int idx = blockIdx.x * 256 + threadIdx.x;     // < BATCH*NCHUNK*DK
  int col = idx & 1023;
  int c   = (idx >> 10) & (NCHUNK - 1);
  int b   = idx >> 15;
  size_t base = ((size_t)b * SEQ_T + c * CHUNK) * DK + col;
  float s = 0.0f;
  for (int t = 0; t < CHUNK; ++t) {
    s += g[base + (size_t)t * DK];
    g[base + (size_t)t * DK] = s;
  }
}

// ---------------------------------------------------------------------------
// GLA chunk-scan kernel.  One block per (batch, head, 64-wide V tile).
// State S[256][64] f32 in LDS across 32 chunks + bf16 transposed mirror
// St[64][256] (written during the update phase) so every WMMA operand is a
// pair of ds_load_b128s.  Dynamic LDS:
//   S(64K) + qg(32K) + St(32K) + kbuf(8K) + vbT(8K) + Am(8K) + blast(1K)
// ---------------------------------------------------------------------------
#define GLA_LDS_BYTES (65536 + 32768 + 32768 + 8192 + 8192 + 8192 + 1024)

__global__ __launch_bounds__(256) void gla_kernel(
    const float* __restrict__ q, const float* __restrict__ k,
    const float* __restrict__ v, const float* __restrict__ bcum,
    float* __restrict__ o) {
  extern __shared__ char smem[];
  float*  S     = (float*) (smem);                   // [256][64] f32 master
  __bf16* qg    = (__bf16*)(smem + 65536);           // [64][256] bf16
  __bf16* St    = (__bf16*)(smem + 98304);           // [64(v)][256(d)] mirror
  __bf16* kbuf  = (__bf16*)(smem + 131072);          // [64][64] staging
  __bf16* vbT   = (__bf16*)(smem + 139264);          // [64(v)][64(t)]
  __bf16* Am    = (__bf16*)(smem + 147456);          // [64][64] masked scores
  float*  blast = (float*) (smem + 155648);          // [256]

  const int tid  = threadIdx.x;
  const int lane = tid & 31, wave = tid >> 5;
  const int hi = lane >> 4, ll = lane & 15;
  const int khalf = hi ? 8 : 0, koff = hi ? 16 : 0;
  const int wmt = (wave >> 1) * 16;     // this wave's 16-row tile within 64
  const int wnt = (wave & 1) * 32;      // this wave's two 16-col tiles base
  const int vt = blockIdx.x, h = blockIdx.y, b = blockIdx.z;
  const int dcol = h * HDK;
  const int vcol = h * HDV + vt * 64;
  const float scale = 0.0625f;          // HDK^-0.5

  for (int i = tid; i < 256 * 64; i += 256) S[i] = 0.0f;
  for (int i = tid; i < 64 * 256; i += 256) St[i] = (__bf16)0.0f;
  __syncthreads();

  for (int c = 0; c < NCHUNK; ++c) {
    const size_t trow = (size_t)b * SEQ_T + (size_t)c * CHUNK;

    // ---- stage blast = cumsum at t=63, qg = q*exp(b)*scale, v tile ----
    blast[tid] = bcum[(trow + 63) * DK + dcol + tid];
    for (int i = tid; i < 64 * 256; i += 256) {
      int t = i >> 8, d = i & 255;
      size_t gix = (trow + t) * DK + dcol + d;
      qg[i] = (__bf16)(q[gix] * __expf(bcum[gix]) * scale);
    }
    for (int i = tid; i < 64 * 64; i += 256) {
      int s = i >> 6, vv = i & 63;
      vbT[vv * 64 + s] = (__bf16)(v[(trow + s) * DV + vcol + vv]);
    }
    __syncthreads();

    // ---- scores A[t][s] = qg . kd^T, K = 256 sliced by 64 over d ----
    v8f accS0 = zero8(), accS1 = zero8();
    for (int db = 0; db < 4; ++db) {
      for (int i = tid; i < 64 * 64; i += 256) {
        int t = i >> 6, dd = i & 63;
        size_t gix = (trow + t) * DK + dcol + db * 64 + dd;
        kbuf[t * 64 + dd] = (__bf16)(k[gix] * __expf(-bcum[gix]));  // Bt[n=s][k=d]
      }
      __syncthreads();
#pragma unroll
      for (int ks = 0; ks < 2; ++ks) {
        v16bf a  = ldfragA(&qg[(wmt + ll) * 256 + db * 64 + ks * 32], khalf);
        v16bf b0 = ldfragB(&kbuf[(wnt + ll) * 64 + ks * 32], koff);
        v16bf b1 = ldfragB(&kbuf[(wnt + 16 + ll) * 64 + ks * 32], koff);
        accS0 = wmma_bf16(a, b0, accS0);
        accS1 = wmma_bf16(a, b1, accS1);
      }
      __syncthreads();
    }

    // ---- causal mask in-register, spill masked A to LDS as bf16 ----
#pragma unroll
    for (int r = 0; r < 8; ++r) {
      int m  = wmt + r + hi * 8;
      int n0 = wnt + ll, n1 = wnt + 16 + ll;
      Am[m * 64 + n0] = (__bf16)((m >= n0) ? accS0[r] : 0.0f);
      Am[m * 64 + n1] = (__bf16)((m >= n1) ? accS1[r] : 0.0f);
    }
    __syncthreads();

    // ---- o = Am @ v  +  qg @ S (via bf16 mirror St) ----
    v8f accO0 = zero8(), accO1 = zero8();
#pragma unroll
    for (int ks = 0; ks < 2; ++ks) {        // intra-chunk (K = 64 over s)
      v16bf a  = ldfragA(&Am[(wmt + ll) * 64 + ks * 32], khalf);
      v16bf b0 = ldfragB(&vbT[(wnt + ll) * 64 + ks * 32], koff);
      v16bf b1 = ldfragB(&vbT[(wnt + 16 + ll) * 64 + ks * 32], koff);
      accO0 = wmma_bf16(a, b0, accO0);
      accO1 = wmma_bf16(a, b1, accO1);
    }
#pragma unroll
    for (int ks = 0; ks < 8; ++ks) {        // inter-chunk (K = 256 over d)
      v16bf a  = ldfragA(&qg[(wmt + ll) * 256 + ks * 32], khalf);
      v16bf b0 = ldfragB(&St[(wnt + ll) * 256 + ks * 32], koff);
      v16bf b1 = ldfragB(&St[(wnt + 16 + ll) * 256 + ks * 32], koff);
      accO0 = wmma_bf16(a, b0, accO0);
      accO1 = wmma_bf16(a, b1, accO1);
    }
#pragma unroll
    for (int r = 0; r < 8; ++r) {
      int m = wmt + r + hi * 8;
      o[(trow + m) * DV + vcol + wnt + ll]      = accO0[r];
      o[(trow + m) * DV + vcol + wnt + 16 + ll] = accO1[r];
    }
    __syncthreads();   // all reads of old S/St and chunk tiles complete

    // ---- S = S * exp(b_last)  +  kg^T @ v ;  refresh bf16 mirror ----
    for (int i = tid; i < 256 * 64; i += 256) S[i] *= __expf(blast[i >> 6]);
    __syncthreads();
    for (int db = 0; db < 4; ++db) {
      for (int i = tid; i < 64 * 64; i += 256) {     // kg^T slice [d][t]
        int t = i >> 6, dd = i & 63;
        size_t gix = (trow + t) * DK + dcol + db * 64 + dd;
        kbuf[dd * 64 + t] = (__bf16)(k[gix] * __expf(blast[db * 64 + dd] - bcum[gix]));
      }
      __syncthreads();
      v8f u0, u1;
#pragma unroll
      for (int r = 0; r < 8; ++r) {                  // C = scaled S tiles
        int d = db * 64 + wmt + r + hi * 8;
        u0[r] = S[d * 64 + wnt + ll];
        u1[r] = S[d * 64 + wnt + 16 + ll];
      }
#pragma unroll
      for (int ks = 0; ks < 2; ++ks) {
        v16bf a  = ldfragA(&kbuf[(wmt + ll) * 64 + ks * 32], khalf);
        v16bf b0 = ldfragB(&vbT[(wnt + ll) * 64 + ks * 32], koff);
        v16bf b1 = ldfragB(&vbT[(wnt + 16 + ll) * 64 + ks * 32], koff);
        u0 = wmma_bf16(a, b0, u0);
        u1 = wmma_bf16(a, b1, u1);
      }
#pragma unroll
      for (int r = 0; r < 8; ++r) {
        int d = db * 64 + wmt + r + hi * 8;
        S[d * 64 + wnt + ll]       = u0[r];
        S[d * 64 + wnt + 16 + ll]  = u1[r];
        St[(wnt + ll) * 256 + d]      = (__bf16)u0[r];   // bf16 mirror [v][d]
        St[(wnt + 16 + ll) * 256 + d] = (__bf16)u1[r];
      }
      __syncthreads();
    }
  }
}

// ---------------------------------------------------------------------------
// RMS-norm over each head's 512 values, times rms_w, times SiLU(x@Wg) -> bf16
// ---------------------------------------------------------------------------
__global__ __launch_bounds__(256) void rms_silu_kernel(
    const float* __restrict__ o, const float* __restrict__ xg,
    const float* __restrict__ rms_w, __bf16* __restrict__ o2) {
  __shared__ float red[256];
  const int tid = threadIdx.x;
  const size_t m = blockIdx.x;
  const int h = blockIdx.y;
  const float* op = o  + m * DV + h * HDV;
  const float* gp = xg + m * DV + h * HDV;
  __bf16*      dp = o2 + m * DV + h * HDV;

  float a0 = op[tid], a1 = op[tid + 256];
  red[tid] = a0 * a0 + a1 * a1;
  __syncthreads();
  for (int st = 128; st > 0; st >>= 1) {
    if (tid < st) red[tid] += red[tid + st];
    __syncthreads();
  }
  float inv = rsqrtf(red[0] * (1.0f / HDV) + EPS_RMS);
  for (int j = tid; j < HDV; j += 256) {
    float z = gp[j];
    float sil = z / (1.0f + __expf(-z));
    dp[j] = (__bf16)(op[j] * inv * rms_w[j] * sil);
  }
}

// ---------------------------------------------------------------------------
// Host-side orchestration (all launches on `stream`; ws usage ~194 MB).
// ---------------------------------------------------------------------------
extern "C" void kernel_launch(void* const* d_in, const int* in_sizes, int n_in,
                              void* d_out, int out_size, void* d_ws, size_t ws_size,
                              hipStream_t stream) {
  const float* x     = (const float*)d_in[0];
  const float* Wq    = (const float*)d_in[1];
  const float* Wk    = (const float*)d_in[2];
  const float* Wv    = (const float*)d_in[3];
  const float* Wg    = (const float*)d_in[4];
  const float* Wgk1  = (const float*)d_in[5];
  const float* Wgk2  = (const float*)d_in[6];
  const float* bgk2  = (const float*)d_in[7];
  const float* Wo    = (const float*)d_in[8];
  const float* rms_w = (const float*)d_in[9];
  float* out = (float*)d_out;

  char* w = (char*)d_ws;
  size_t off = 0;
  auto nxt = [&](size_t bytes) -> char* {
    char* p = w + off;
    off += (bytes + 255) & ~(size_t)255;
    return p;
  };
  __bf16* xbf  = (__bf16*)nxt((size_t)NTOK * D_MODEL * 2);
  __bf16* wqbf = (__bf16*)nxt((size_t)D_MODEL * DK * 2);
  __bf16* wkbf = (__bf16*)nxt((size_t)D_MODEL * DK * 2);
  __bf16* wvbf = (__bf16*)nxt((size_t)D_MODEL * DV * 2);
  __bf16* wgbf = (__bf16*)nxt((size_t)D_MODEL * DV * 2);
  __bf16* wobf = (__bf16*)nxt((size_t)DV * D_MODEL * 2);
  float*  qf   = (float*) nxt((size_t)NTOK * DK * 4);
  float*  kf   = (float*) nxt((size_t)NTOK * DK * 4);
  float*  vf   = (float*) nxt((size_t)NTOK * DV * 4);
  float*  xgf  = (float*) nxt((size_t)NTOK * DV * 4);
  float*  t1f  = (float*) nxt((size_t)NTOK * LOW_RANK * 4);
  float*  gf   = (float*) nxt((size_t)NTOK * DK * 4);     // g, then in-place cumsum
  float*  of   = (float*) nxt((size_t)NTOK * DV * 4);
  __bf16* o2bf = (__bf16*)nxt((size_t)NTOK * DV * 2);
  (void)ws_size; (void)in_sizes; (void)n_in; (void)out_size;

  // 1) bf16 casts
  cast_kernel<<<(NTOK * D_MODEL) / 256, 256, 0, stream>>>(x,  xbf,  NTOK * D_MODEL);
  cast_kernel<<<(D_MODEL * DK) / 256, 256, 0, stream>>>(Wq, wqbf, D_MODEL * DK);
  cast_kernel<<<(D_MODEL * DK) / 256, 256, 0, stream>>>(Wk, wkbf, D_MODEL * DK);
  cast_kernel<<<(D_MODEL * DV) / 256, 256, 0, stream>>>(Wv, wvbf, D_MODEL * DV);
  cast_kernel<<<(D_MODEL * DV) / 256, 256, 0, stream>>>(Wg, wgbf, D_MODEL * DV);
  cast_kernel<<<(DV * D_MODEL) / 256, 256, 0, stream>>>(Wo, wobf, DV * D_MODEL);

  // 2) projections (WMMA GEMMs)
  gemm_bf16_kernel<<<dim3(DK / 128, NTOK / 128), 256, 0, stream>>>(xbf, wqbf, qf,  NTOK, DK, D_MODEL);
  gemm_bf16_kernel<<<dim3(DK / 128, NTOK / 128), 256, 0, stream>>>(xbf, wkbf, kf,  NTOK, DK, D_MODEL);
  gemm_bf16_kernel<<<dim3(DV / 128, NTOK / 128), 256, 0, stream>>>(xbf, wvbf, vf,  NTOK, DV, D_MODEL);
  gemm_bf16_kernel<<<dim3(DV / 128, NTOK / 128), 256, 0, stream>>>(xbf, wgbf, xgf, NTOK, DV, D_MODEL);

  // 3) low-rank gates + per-chunk cumsum
  lr1_kernel<<<(NTOK * LOW_RANK) / 256, 256, 0, stream>>>(x, Wgk1, t1f);
  gate_kernel<<<(NTOK * DK) / 256, 256, 0, stream>>>(t1f, Wgk2, bgk2, gf);
  cumsum_kernel<<<(BATCH * NCHUNK * DK) / 256, 256, 0, stream>>>(gf);

  // 4) gated linear attention scan
  gla_kernel<<<dim3(HDV / 64, HEADS, BATCH), 256, GLA_LDS_BYTES, stream>>>(qf, kf, vf, gf, of);

  // 5) RMS norm * SiLU gate -> bf16
  rms_silu_kernel<<<dim3(NTOK, HEADS), 256, 0, stream>>>(of, xgf, rms_w, o2bf);

  // 6) output projection
  gemm_bf16_kernel<<<dim3(D_MODEL / 128, NTOK / 128), 256, 0, stream>>>(o2bf, wobf, out, NTOK, D_MODEL, DV);
}
